// VectorQuantize_69252052680739
// MI455X (gfx1250) — matrix operand order repaired and checked by reading, not compiled, not run on hardware
//
#include <hip/hip_runtime.h>
#include <hip/hip_bf16.h>

typedef __attribute__((ext_vector_type(16))) _Float16 v16h;
typedef __attribute__((ext_vector_type(8)))  _Float16 v8h;
typedef __attribute__((ext_vector_type(8)))  float    v8f;

#define N_PTS 65536
#define DIM   64
#define KCB   2048
#define LDSE_STRIDE 72   // halves per codebook row in LDS: 144B (16B aligned), bank-conflict-free

// ---------------- K0: convert codebook f32->f16, compute e2[k] ----------------
__global__ void vq_prep(const float* __restrict__ embed,
                        _Float16* __restrict__ embed_h,
                        float* __restrict__ e2) {
  int wave = threadIdx.x >> 5, lane = threadIdx.x & 31;
  int k = blockIdx.x * 8 + wave;
  const float* row = embed + k * DIM;
  float a = row[lane * 2], b = row[lane * 2 + 1];
  embed_h[k * DIM + lane * 2]     = (_Float16)a;
  embed_h[k * DIM + lane * 2 + 1] = (_Float16)b;
  float s = a * a + b * b;
  for (int m = 16; m >= 1; m >>= 1) s += __shfl_xor(s, m, 32);
  if (lane == 0) e2[k] = s;
}

__device__ inline void cvt8(v16h& d, int off, float4 a, float4 b) {
  d[off + 0] = (_Float16)a.x; d[off + 1] = (_Float16)a.y;
  d[off + 2] = (_Float16)a.z; d[off + 3] = (_Float16)a.w;
  d[off + 4] = (_Float16)b.x; d[off + 5] = (_Float16)b.y;
  d[off + 6] = (_Float16)b.z; d[off + 7] = (_Float16)b.w;
}

// Build A operand pair (K=0..31 and K=32..63) for one 16-row tile starting at row m.
// Layout (ISA 7.12.2, 16-bit A 16x32): lane<16 -> K {0..7,16..23}; lane>=16 -> K {8..15,24..31}
__device__ inline void load_a(const float* __restrict__ x, int m, int hsel,
                              v16h& a0, v16h& a1) {
  const float4* xr = (const float4*)(x + (size_t)m * DIM);
  int c1 = hsel * 2;                                   // float4 index of K base
  cvt8(a0, 0, xr[c1],      xr[c1 + 1]);
  cvt8(a0, 8, xr[c1 + 4],  xr[c1 + 5]);
  cvt8(a1, 0, xr[c1 + 8],  xr[c1 + 9]);
  cvt8(a1, 8, xr[c1 + 12], xr[c1 + 13]);
}

// ---------------- K1: fused distance-GEMM + argmin (WMMA f16->f32) ----------------
// 8 waves/block; wave owns 32 rows (2 A-operand sets) so each B tile fetched from
// LDS feeds 4 WMMAs -> halves LDS bytes per FLOP. Whole f16 codebook LDS-resident.
__global__ void __launch_bounds__(256) vq_assign(const float* __restrict__ x,
    const _Float16* __restrict__ embed_h, const float* __restrict__ e2,
    int* __restrict__ idx_out, float* __restrict__ ind_f_out) {
  __shared__ _Float16 lds_e[KCB * LDSE_STRIDE];   // 288 KB
  __shared__ float    lds_e2[KCB];                //   8 KB

  int tid = threadIdx.x;
  // Stage codebook: 2048 rows x 64 halves, padded stride 72
  for (int c = tid; c < KCB * 8; c += 256) {      // 8 x v8h chunks per row
    int row = c >> 3, c8 = c & 7;
    v8h v = *(const v8h*)(embed_h + row * DIM + c8 * 8);
    *(v8h*)(&lds_e[row * LDSE_STRIDE + c8 * 8]) = v;
  }
  for (int c = tid; c < KCB; c += 256) lds_e2[c] = e2[c];
  __syncthreads();

  int wave = tid >> 5, lane = tid & 31;
  int hsel = lane >> 4;                 // lane half selects K sub-block
  int row_base = blockIdx.x * 256 + wave * 32;

  v16h a0, a1, a2, a3;
  load_a(x, row_base + (lane & 15),      hsel, a0, a1);   // rows [row_base, +16)
  load_a(x, row_base + 16 + (lane & 15), hsel, a2, a3);   // rows [row_base+16, +32)

  float bv0[8], bv1[8]; int bi0[8], bi1[8];
#pragma unroll
  for (int r = 0; r < 8; ++r) {
    bv0[r] = 3.4e38f; bi0[r] = 0;
    bv1[r] = 3.4e38f; bi1[r] = 0;
  }

#pragma unroll 2
  for (int t = 0; t < KCB / 16; ++t) {
    int e = t * 16 + (lane & 15);                        // codebook entry (B column)
    // B operand (32x16 KxN): lanes 0-15 hold K {0..7,16..23}, lanes 16-31 hold K {8..15,24..31}
    const _Float16* rp = &lds_e[e * LDSE_STRIDE + hsel * 8];
    v8h b00 = *(const v8h*)(rp);
    v8h b01 = *(const v8h*)(rp + 16);
    v8h b10 = *(const v8h*)(rp + 32);
    v8h b11 = *(const v8h*)(rp + 48);
    v16h b0 = __builtin_shufflevector(b00, b01, 0,1,2,3,4,5,6,7,8,9,10,11,12,13,14,15);
    v16h b1 = __builtin_shufflevector(b10, b11, 0,1,2,3,4,5,6,7,8,9,10,11,12,13,14,15);

    v8f c0 = {}, c1 = {};
    c0 = __builtin_amdgcn_wmma_f32_16x16x32_f16(false, a0, false, b0, (short)0, c0, false, false);
    c1 = __builtin_amdgcn_wmma_f32_16x16x32_f16(false, a2, false, b0, (short)0, c1, false, false);
    c0 = __builtin_amdgcn_wmma_f32_16x16x32_f16(false, a1, false, b1, (short)0, c0, false, false);
    c1 = __builtin_amdgcn_wmma_f32_16x16x32_f16(false, a3, false, b1, (short)0, c1, false, false);

    float ev = lds_e2[e];
#pragma unroll
    for (int r = 0; r < 8; ++r) {                        // dist = e2 - 2*dot (x2 row-const)
      float d0 = ev - 2.0f * c0[r];
      if (d0 < bv0[r]) { bv0[r] = d0; bi0[r] = e; }
      float d1 = ev - 2.0f * c1[r];
      if (d1 < bv1[r]) { bv1[r] = d1; bi1[r] = e; }
    }
  }

  // Cross-lane argmin within each 16-lane half; C-layout: c[r] is row r+8*hsel, col lane&15
#pragma unroll
  for (int r = 0; r < 8; ++r) {
    float v0 = bv0[r]; int i0 = bi0[r];
    float v1 = bv1[r]; int i1 = bi1[r];
#pragma unroll
    for (int mm = 8; mm >= 1; mm >>= 1) {
      float ov0 = __shfl_xor(v0, mm, 32);
      int   oi0 = __shfl_xor(i0, mm, 32);
      if (ov0 < v0 || (ov0 == v0 && oi0 < i0)) { v0 = ov0; i0 = oi0; }
      float ov1 = __shfl_xor(v1, mm, 32);
      int   oi1 = __shfl_xor(i1, mm, 32);
      if (ov1 < v1 || (ov1 == v1 && oi1 < i1)) { v1 = ov1; i1 = oi1; }
    }
    if ((lane & 15) == 0) {
      int row0 = row_base + r + 8 * hsel;
      idx_out[row0]   = i0;
      ind_f_out[row0] = (float)i0;
      int row1 = row0 + 16;
      idx_out[row1]   = i1;
      ind_f_out[row1] = (float)i1;
    }
  }
}

// ---------------- K2: gather quantize, commit loss, segment-sum scatter ----------------
__global__ void vq_gather_scatter(const float* __restrict__ x, const float* __restrict__ embed,
    const int* __restrict__ idx, float* __restrict__ q_out,
    float* __restrict__ counts, float* __restrict__ esum, float* __restrict__ loss) {
  __shared__ float red[8];
  int g = blockIdx.x * 256 + threadIdx.x;
  int row = g >> 6, d = g & 63;
  int k = idx[row];
  float xv = x[g];
  float q  = embed[k * DIM + d];
  q_out[g] = q;                               // quantize_st == quantize numerically
  float df = q - xv;
  float sq = df * df;
  atomicAdd(&esum[k * DIM + d], xv);
  if (d == 0) atomicAdd(&counts[k], 1.0f);
  for (int m = 16; m >= 1; m >>= 1) sq += __shfl_xor(sq, m, 32);
  int lane = threadIdx.x & 31, w = threadIdx.x >> 5;
  if (lane == 0) red[w] = sq;
  __syncthreads();
  if (threadIdx.x == 0) {
    float s = 0.f;
    for (int i = 0; i < 8; ++i) s += red[i];
    atomicAdd(loss, s);
  }
}

// ---------------- K3a: laplace smoothing (single block) + loss finalize ----------------
__global__ void vq_smooth(const float* __restrict__ cluster_size, const float* __restrict__ counts,
    const float* __restrict__ loss, float* __restrict__ smoothed, float* __restrict__ loss_out) {
  __shared__ float red[8];
  __shared__ float n_sh;
  float part = 0.f;
  for (int k = threadIdx.x; k < KCB; k += 256) {
    float ncs = 0.99f * cluster_size[k] + 0.01f * counts[k];
    smoothed[k] = ncs;
    part += ncs;
  }
  for (int m = 16; m >= 1; m >>= 1) part += __shfl_xor(part, m, 32);
  int lane = threadIdx.x & 31, w = threadIdx.x >> 5;
  if (lane == 0) red[w] = part;
  __syncthreads();
  if (threadIdx.x == 0) {
    float n = 0.f;
    for (int i = 0; i < 8; ++i) n += red[i];
    n_sh = n;
    loss_out[0] = 1.0f * loss[0] / (float)((size_t)N_PTS * DIM);   // COMMIT_W * mean
  }
  __syncthreads();
  float n = n_sh;
  for (int k = threadIdx.x; k < KCB; k += 256) {
    float ncs = smoothed[k];
    smoothed[k] = (ncs + 1e-5f) / (n + KCB * 1e-5f) * n;
  }
}

// ---------------- K3b: new_embed = EMA(embed_avg)/smoothed ----------------
__global__ void vq_newembed(const float* __restrict__ embed_avg, const float* __restrict__ esum,
    const float* __restrict__ smoothed, float* __restrict__ out) {
  int g = blockIdx.x * 256 + threadIdx.x;
  float na = 0.99f * embed_avg[g] + 0.01f * esum[g];
  out[g] = na / smoothed[g >> 6];
}

extern "C" void kernel_launch(void* const* d_in, const int* in_sizes, int n_in,
                              void* d_out, int out_size, void* d_ws, size_t ws_size,
                              hipStream_t stream) {
  const float* x            = (const float*)d_in[0];
  const float* embed        = (const float*)d_in[1];
  const float* embed_avg    = (const float*)d_in[2];
  const float* cluster_size = (const float*)d_in[3];
  float* out = (float*)d_out;
  char*  ws  = (char*)d_ws;

  // workspace layout (bytes)
  _Float16* ws_eh    = (_Float16*)(ws + 0);        // 262144
  float*    ws_e2    = (float*)(ws + 262144);      //   8192
  int*      ws_idx   = (int*)  (ws + 270336);      // 262144
  float*    ws_counts= (float*)(ws + 532480);      //   8192
  float*    ws_esum  = (float*)(ws + 540672);      // 524288
  float*    ws_loss  = (float*)(ws + 1064960);     //      4 (pad 64)
  float*    ws_sm    = (float*)(ws + 1065024);     //   8192

  // zero counts + esum + loss each call (graph-capture-safe stream op)
  hipMemsetAsync(ws + 532480, 0, 532544, stream);

  // d_out layout: quantize[N*D] | loss[1] | embed_ind[N] | new_embed[K*D]
  float* out_q    = out;
  float* out_loss = out + (size_t)N_PTS * DIM;
  float* out_ind  = out + (size_t)N_PTS * DIM + 1;
  float* out_emb  = out + (size_t)N_PTS * DIM + 1 + N_PTS;

  vq_prep<<<KCB / 8, 256, 0, stream>>>(embed, ws_eh, ws_e2);
  vq_assign<<<N_PTS / 256, 256, 0, stream>>>(x, ws_eh, ws_e2, ws_idx, out_ind);
  vq_gather_scatter<<<(N_PTS * DIM) / 256, 256, 0, stream>>>(x, embed, ws_idx, out_q,
                                                             ws_counts, ws_esum, ws_loss);
  vq_smooth<<<1, 256, 0, stream>>>(cluster_size, ws_counts, ws_loss, ws_sm, out_loss);
  vq_newembed<<<(KCB * DIM) / 256, 256, 0, stream>>>(embed_avg, ws_esum, ws_sm, out_emb);
}
